// Attention_25031069401226
// MI455X (gfx1250) — matrix operand order repaired and checked by reading, not compiled, not run on hardware
//
#include <hip/hip_runtime.h>

constexpr int BATCH  = 2;
constexpr int SEQ    = 2048;
constexpr int HEADS  = 16;
constexpr int DMODEL = 1024;
constexpr int DHEAD  = 64;
constexpr int TOK    = BATCH * SEQ;   // 4096

typedef __attribute__((ext_vector_type(16))) __bf16 v16bf;
typedef __attribute__((ext_vector_type(8)))  float  v8f;
typedef unsigned int u32x4 __attribute__((ext_vector_type(4)));
typedef int          i32x4 __attribute__((ext_vector_type(4)));
typedef int          i32x8 __attribute__((ext_vector_type(8)));

union ABFrag { uint4 q[2]; v16bf v; };

#if defined(__has_builtin)
#if __has_builtin(__builtin_amdgcn_tensor_load_to_lds) && __has_builtin(__builtin_amdgcn_s_wait_tensorcnt)
#define HAVE_TDM 1
#else
#define HAVE_TDM 0
#endif
#else
#define HAVE_TDM 0
#endif

#if defined(__has_include)
#if __has_include(<hip/amd_detail/amd_gfx1250_TDM.h>)
#define TDM_SIX_ARGS 1
#else
#define TDM_SIX_ARGS 0
#endif
#else
#define TDM_SIX_ARGS 0
#endif

__device__ __forceinline__ unsigned short f2bf(float f) {
  union { float f; unsigned int u; } x; x.f = f;
  unsigned int r = x.u + 0x7FFFu + ((x.u >> 16) & 1u);
  return (unsigned short)(r >> 16);
}

__device__ __forceinline__ v8f wmma_bf16(const ABFrag& a, const ABFrag& b, v8f c) {
  return __builtin_amdgcn_wmma_f32_16x16x32_bf16(false, a.v, false, b.v, (short)0, c,
                                                 false, false);
}

__device__ __forceinline__ unsigned lds_off(const void* p) {
  // generic pointer to LDS: low 32 bits are the workgroup-relative LDS offset
  return (unsigned)(unsigned long long)(uintptr_t)p;
}

#if HAVE_TDM
// TDM 2D tile load: bf16 elements (data_size=1 -> 2 bytes), tile (tile_w x tile_h),
// row stride stride0 (elements).  Tensor dims = tile dims (tile always in bounds).
// D# bit packing per CDNA5 ISA section 8 (group0: count/lds/global/type, group1:
// data_size, dims, tile dims, stride).
__device__ __forceinline__ void tdm_load_2d(unsigned lds_addr, const void* gsrc,
                                            unsigned tile_w, unsigned tile_h,
                                            unsigned stride0) {
  unsigned long long ga = (unsigned long long)(uintptr_t)gsrc;
  u32x4 g0;
  g0[0] = 1u;                                             // count=1 (valid user D#)
  g0[1] = lds_addr;                                       // lds_addr [63:32]
  g0[2] = (unsigned)(ga & 0xFFFFFFFFull);                 // global_addr[31:0]
  g0[3] = (unsigned)((ga >> 32) & 0x1FFFFFFull) | (2u << 30);  // addr[56:32] | type=2
  i32x8 g1;
  g1[0] = (int)(1u << 16);                                // data_size=1 (2B), mask=0
  g1[1] = (int)((tile_w & 0xFFFFu) << 16);                // tensor_dim0[15:0]
  g1[2] = (int)((tile_w >> 16) | ((tile_h & 0xFFFFu) << 16));   // dim0 hi | dim1 lo
  g1[3] = (int)((tile_h >> 16) | ((tile_w & 0xFFFFu) << 16));   // dim1 hi | tile_dim0
  g1[4] = (int)(tile_h & 0xFFFFu);                        // tile_dim1 | tile_dim2=0
  g1[5] = (int)stride0;                                   // tensor_dim0_stride[31:0]
  g1[6] = 0;                                              // stride0 hi | stride1 lo
  g1[7] = 0;
  i32x4 z4 = {0, 0, 0, 0};
#if TDM_SIX_ARGS
  i32x8 z8 = {0, 0, 0, 0, 0, 0, 0, 0};
  __builtin_amdgcn_tensor_load_to_lds(g0, g1, z4, z4, z8, 0);
#else
  __builtin_amdgcn_tensor_load_to_lds(g0, g1, z4, z4, 0);
#endif
}
#endif

// ---------------------------------------------------------------------------
// Stage 0: conversions fp32 -> bf16 (with transposes for weight B-operands)
// ---------------------------------------------------------------------------
__global__ void cvt_act(const float* __restrict__ s, unsigned short* __restrict__ d) {
  int i = blockIdx.x * 256 + threadIdx.x;
  d[i] = f2bf(s[i]);
}

__global__ void cvt_wqkv_t(const float* __restrict__ s, unsigned short* __restrict__ d) {
  int i  = blockIdx.x * 256 + threadIdx.x;         // over H*DMODEL*DHEAD
  int dd = i % DHEAD;
  int m  = (i / DHEAD) % DMODEL;
  int h  = i / (DHEAD * DMODEL);
  d[(size_t)(h * DHEAD + dd) * DMODEL + m] = f2bf(s[i]);
}

__global__ void cvt_wo_t(const float* __restrict__ s, unsigned short* __restrict__ d) {
  int i  = blockIdx.x * 256 + threadIdx.x;         // over H*DHEAD*DMODEL
  int m  = i % DMODEL;
  int hd = i / DMODEL;
  d[(size_t)m * (HEADS * DHEAD) + hd] = f2bf(s[i]);
}

// ---------------------------------------------------------------------------
// Shared GEMM fragment loaders (A: 4 row-subtiles, B: 4 col-subtiles)
// ---------------------------------------------------------------------------
__device__ __forceinline__ void load_a4(ABFrag a[4], const uint4* const* Xr, int jb,
                                        int half) {
#pragma unroll
  for (int s = 0; s < 4; ++s) {
    a[s].q[0] = Xr[s][jb + half];
    a[s].q[1] = Xr[s][jb + 2 + half];
  }
}
__device__ __forceinline__ void load_b4(ABFrag b[4], const uint4* const* Wr, int jb,
                                        int half) {
#pragma unroll
  for (int f = 0; f < 4; ++f) {
    b[f].q[0] = Wr[f][jb + half];
    b[f].q[1] = Wr[f][jb + 2 + half];
  }
}

// ---------------------------------------------------------------------------
// Stage 1: QKV projections.  One wave -> 64-token x 64-col tile, double-buffered.
// ---------------------------------------------------------------------------
__global__ __launch_bounds__(32) void qkv_proj(
    const unsigned short* __restrict__ Xq, const unsigned short* __restrict__ Xk,
    const unsigned short* __restrict__ Xv,
    const unsigned short* __restrict__ WqT, const unsigned short* __restrict__ WkT,
    const unsigned short* __restrict__ WvT,
    const float* __restrict__ bQ, const float* __restrict__ bK, const float* __restrict__ bV,
    unsigned short* __restrict__ Qo, unsigned short* __restrict__ Ko,
    unsigned short* __restrict__ VtO) {
  const int lane = threadIdx.x;
  const int half = lane >> 4;
  const int l16  = lane & 15;
  const int mt   = blockIdx.x;     // 64-token tile
  const int nt   = blockIdx.y;     // 64-column tile
  const int which = blockIdx.z;    // 0=Q 1=K 2=V

  const unsigned short* X = (which == 0) ? Xq : (which == 1) ? Xk : Xv;
  const unsigned short* W = (which == 0) ? WqT : (which == 1) ? WkT : WvT;
  const float* bias       = (which == 0) ? bQ : (which == 1) ? bK : bV;

  const v8f z8 = {0.f, 0.f, 0.f, 0.f, 0.f, 0.f, 0.f, 0.f};
  v8f acc[4][4];
#pragma unroll
  for (int s = 0; s < 4; ++s)
#pragma unroll
    for (int f = 0; f < 4; ++f) acc[s][f] = z8;

  const uint4* Xr[4];
  const uint4* Wr[4];
#pragma unroll
  for (int s = 0; s < 4; ++s)
    Xr[s] = (const uint4*)(X + (size_t)(mt * 64 + s * 16 + l16) * DMODEL);
#pragma unroll
  for (int f = 0; f < 4; ++f)
    Wr[f] = (const uint4*)(W + (size_t)(nt * 64 + f * 16 + l16) * DMODEL);

  ABFrag a[4], bm[4];
  load_a4(a, Xr, 0, half);
  load_b4(bm, Wr, 0, half);
  for (int k0 = 0; k0 < DMODEL; k0 += 32) {
    ABFrag an[4], bn[4];
    const bool more = (k0 + 32 < DMODEL);
    if (more) {
      const int jb = (k0 + 32) >> 3;
      __builtin_prefetch((const void*)(Xr[0] + jb + 8), 0, 0);
      load_a4(an, Xr, jb, half);
      load_b4(bn, Wr, jb, half);
    }
#pragma unroll
    for (int s = 0; s < 4; ++s)
#pragma unroll
      for (int f = 0; f < 4; ++f) acc[s][f] = wmma_bf16(a[s], bm[f], acc[s][f]);
    if (more) {
#pragma unroll
      for (int i = 0; i < 4; ++i) { a[i] = an[i]; bm[i] = bn[i]; }
    }
  }

#pragma unroll
  for (int f = 0; f < 4; ++f) {
    const int n  = nt * 64 + f * 16 + l16;
    const int h  = n >> 6;
    const int dd = n & 63;
    const float bv = bias[n];
#pragma unroll
    for (int s = 0; s < 4; ++s)
#pragma unroll
      for (int g = 0; g < 8; ++g) {
        const int token = mt * 64 + s * 16 + g + 8 * half;   // C row = g + 8*half
        const int b  = token / SEQ;
        const int sq = token % SEQ;
        const float val = acc[s][f][g] + bv;
        if (which == 2)
          VtO[(size_t)((b * HEADS + h) * DHEAD + dd) * SEQ + sq] = f2bf(val);
        else if (which == 0)
          Qo[(size_t)((b * HEADS + h) * SEQ + sq) * DHEAD + dd] = f2bf(val);
        else
          Ko[(size_t)((b * HEADS + h) * SEQ + sq) * DHEAD + dd] = f2bf(val);
      }
  }
}

// ---------------------------------------------------------------------------
// Stage 2: causal flash attention.  One wave per (b, h, 32-query tile).
// K/V tiles staged into LDS by the Tensor Data Mover, double-buffered.
// ---------------------------------------------------------------------------
__global__ __launch_bounds__(32) void flash_attn(
    const unsigned short* __restrict__ Q, const unsigned short* __restrict__ K,
    const unsigned short* __restrict__ Vt, const float* __restrict__ mask,
    unsigned short* __restrict__ Z) {
  const int lane = threadIdx.x;
  const int half = lane >> 4;
  const int l16  = lane & 15;
  const int qt   = blockIdx.x;          // 32-query tile
  const int h    = blockIdx.y;
  const int b    = blockIdx.z;

  const unsigned short* Qb = Q  + (size_t)(b * HEADS + h) * SEQ * DHEAD;
  const unsigned short* Kb = K  + (size_t)(b * HEADS + h) * SEQ * DHEAD;
  const unsigned short* Vb = Vt + (size_t)(b * HEADS + h) * DHEAD * SEQ;
  const float* mp = mask + (size_t)b * SEQ;

  __shared__ __align__(16) unsigned short KT[2][32 * 64];  // 2 x 4 KiB
  __shared__ __align__(16) unsigned short VT[2][64 * 32];  // 2 x 4 KiB
  __shared__ __align__(16) float          Sc[32][32];
  __shared__ __align__(16) unsigned short Pt[32][32];
  __shared__ float Alpha[32];
  __shared__ float InvL[32];
  __shared__ float Mk[32];

  // loop-invariant Q A-fragments: 2 row-subtiles x 2 k-steps (dh 0..31 / 32..63)
  ABFrag qa[2][2];
#pragma unroll
  for (int s = 0; s < 2; ++s) {
    const uint4* Qr = (const uint4*)(Qb + (size_t)(qt * 32 + s * 16 + l16) * DHEAD);
    qa[s][0].q[0] = Qr[half];      qa[s][0].q[1] = Qr[2 + half];
    qa[s][1].q[0] = Qr[4 + half];  qa[s][1].q[1] = Qr[6 + half];
  }

  const v8f z8 = {0.f, 0.f, 0.f, 0.f, 0.f, 0.f, 0.f, 0.f};
  v8f acc[2][4];
#pragma unroll
  for (int s = 0; s < 2; ++s)
#pragma unroll
    for (int f = 0; f < 4; ++f) acc[s][f] = z8;

  float m_i = -1e30f, l_i = 0.f;        // one softmax row per lane (row = lane)
  const float scale = 0.125f;           // 1/sqrt(64)
  const int qrow = qt * 32 + lane;
  const int nkb = qt + 1;               // causal: keys 0 .. qt*32+31

#if HAVE_TDM
  // prologue: TDM-load key block 0 into buffer 0
  tdm_load_2d(lds_off(&KT[0][0]), Kb, 64, 32, 64u);
  tdm_load_2d(lds_off(&VT[0][0]), Vb, 32, 64, (unsigned)SEQ);
#endif

  for (int kb = 0; kb < nkb; ++kb) {
    const int kbase = kb * 32;
    const int buf = kb & 1;
#if HAVE_TDM
    if (kb + 1 < nkb) {
      const int kn = (kb + 1) * 32;
      tdm_load_2d(lds_off(&KT[buf ^ 1][0]), Kb + (size_t)kn * DHEAD, 64, 32, 64u);
      tdm_load_2d(lds_off(&VT[buf ^ 1][0]), Vb + kn, 32, 64, (unsigned)SEQ);
      __builtin_amdgcn_s_wait_tensorcnt(2);  // in-order: block kb's 2 loads done
    } else {
      __builtin_amdgcn_s_wait_tensorcnt(0);
    }
#else
    // fallback staging: wave copies the 4 KiB tiles through VGPRs
    {
      const uint4* src = (const uint4*)(Kb + (size_t)kbase * DHEAD);
      uint4* dst = (uint4*)&KT[buf][0];
#pragma unroll
      for (int j = 0; j < 8; ++j) dst[lane + 32 * j] = src[lane + 32 * j];
      uint4* dv = (uint4*)&VT[buf][0];
#pragma unroll
      for (int j = 0; j < 8; ++j) {
        const int idx = lane + 32 * j;
        const int row = idx >> 2, c4 = idx & 3;
        dv[idx] = ((const uint4*)(Vb + (size_t)row * SEQ + kbase))[c4];
      }
    }
    __syncthreads();
#endif

    // ---- scores: 2 query-subtiles x 2 key-ntiles, K read from LDS ----
    v8f c[2][2] = {{z8, z8}, {z8, z8}};
#pragma unroll
    for (int ntile = 0; ntile < 2; ++ntile) {
      const uint4* Kr = (const uint4*)&KT[buf][(ntile * 16 + l16) * DHEAD];
      ABFrag b0, b1;
      b0.q[0] = Kr[half];     b0.q[1] = Kr[2 + half];
      b1.q[0] = Kr[4 + half]; b1.q[1] = Kr[6 + half];
#pragma unroll
      for (int s = 0; s < 2; ++s) {
        c[s][ntile] = wmma_bf16(qa[s][0], b0, c[s][ntile]);
        c[s][ntile] = wmma_bf16(qa[s][1], b1, c[s][ntile]);
      }
    }
    Mk[lane] = mp[kbase + lane];
    __syncthreads();
#pragma unroll
    for (int s = 0; s < 2; ++s)
#pragma unroll
      for (int g = 0; g < 8; ++g) {
        Sc[s * 16 + g + 8 * half][l16]      = c[s][0][g];
        Sc[s * 16 + g + 8 * half][l16 + 16] = c[s][1][g];
      }
    __syncthreads();

    // ---- online softmax: every lane owns one query row ----
    {
      float bm = -1e30f;
      for (int j = 0; j < 32; ++j) {
        const int key = kbase + j;
        float sv = Sc[lane][j] * scale + Mk[j];
        if (key > qrow) sv = -1e30f;       // causal mask
        Sc[lane][j] = sv;
        bm = fmaxf(bm, sv);
      }
      const float nm = fmaxf(m_i, bm);
      const float al = __expf(m_i - nm);
      float sum = 0.f;
      for (int j = 0; j < 32; ++j) {
        const float p = __expf(Sc[lane][j] - nm);
        Pt[lane][j] = f2bf(p);
        sum += p;
      }
      l_i = l_i * al + sum;
      m_i = nm;
      Alpha[lane] = al;
    }
    __syncthreads();

    // ---- rescale accumulators, then acc += P * V (V from LDS) ----
    float ar[2][8];
#pragma unroll
    for (int s = 0; s < 2; ++s)
#pragma unroll
      for (int g = 0; g < 8; ++g) ar[s][g] = Alpha[s * 16 + g + 8 * half];
#pragma unroll
    for (int s = 0; s < 2; ++s)
#pragma unroll
      for (int f = 0; f < 4; ++f)
#pragma unroll
        for (int g = 0; g < 8; ++g) acc[s][f][g] *= ar[s][g];

    ABFrag pa[2];
#pragma unroll
    for (int s = 0; s < 2; ++s) {
      const uint4* Pr = (const uint4*)&Pt[s * 16 + l16][0];
      pa[s].q[0] = Pr[half];
      pa[s].q[1] = Pr[2 + half];
    }
#pragma unroll
    for (int f = 0; f < 4; ++f) {
      const uint4* Vr = (const uint4*)&VT[buf][(f * 16 + l16) * 32];
      ABFrag vb;
      vb.q[0] = Vr[half];
      vb.q[1] = Vr[2 + half];
#pragma unroll
      for (int s = 0; s < 2; ++s) acc[s][f] = wmma_bf16(pa[s], vb, acc[s][f]);
    }
  }

  // ---- epilogue ----
  InvL[lane] = 1.0f / l_i;
  __syncthreads();
#pragma unroll
  for (int s = 0; s < 2; ++s) {
    float il[8];
#pragma unroll
    for (int g = 0; g < 8; ++g) il[g] = InvL[s * 16 + g + 8 * half];
#pragma unroll
    for (int f = 0; f < 4; ++f)
#pragma unroll
      for (int g = 0; g < 8; ++g) {
        const int q = qt * 32 + s * 16 + g + 8 * half;
        Z[(size_t)(b * SEQ + q) * (HEADS * DHEAD) + h * DHEAD + f * 16 + l16] =
            f2bf(acc[s][f][g] * il[g]);
      }
  }
}

// ---------------------------------------------------------------------------
// Stage 3: output projection  out = Z * W_O + b_O  (64x64 tile, double-buffered)
// ---------------------------------------------------------------------------
__global__ __launch_bounds__(32) void out_proj(
    const unsigned short* __restrict__ Zm, const unsigned short* __restrict__ WoT,
    const float* __restrict__ bO, float* __restrict__ out) {
  const int lane = threadIdx.x;
  const int half = lane >> 4;
  const int l16  = lane & 15;
  const int mt   = blockIdx.x;
  const int nt   = blockIdx.y;

  const v8f z8 = {0.f, 0.f, 0.f, 0.f, 0.f, 0.f, 0.f, 0.f};
  v8f acc[4][4];
#pragma unroll
  for (int s = 0; s < 4; ++s)
#pragma unroll
    for (int f = 0; f < 4; ++f) acc[s][f] = z8;

  const uint4* Zr[4];
  const uint4* Wr[4];
#pragma unroll
  for (int s = 0; s < 4; ++s)
    Zr[s] = (const uint4*)(Zm + (size_t)(mt * 64 + s * 16 + l16) * (HEADS * DHEAD));
#pragma unroll
  for (int f = 0; f < 4; ++f)
    Wr[f] = (const uint4*)(WoT + (size_t)(nt * 64 + f * 16 + l16) * (HEADS * DHEAD));

  ABFrag a[4], bm[4];
  load_a4(a, Zr, 0, half);
  load_b4(bm, Wr, 0, half);
  for (int k0 = 0; k0 < HEADS * DHEAD; k0 += 32) {
    ABFrag an[4], bn[4];
    const bool more = (k0 + 32 < HEADS * DHEAD);
    if (more) {
      const int jb = (k0 + 32) >> 3;
      __builtin_prefetch((const void*)(Zr[0] + jb + 8), 0, 0);
      load_a4(an, Zr, jb, half);
      load_b4(bn, Wr, jb, half);
    }
#pragma unroll
    for (int s = 0; s < 4; ++s)
#pragma unroll
      for (int f = 0; f < 4; ++f) acc[s][f] = wmma_bf16(a[s], bm[f], acc[s][f]);
    if (more) {
#pragma unroll
      for (int i = 0; i < 4; ++i) { a[i] = an[i]; bm[i] = bn[i]; }
    }
  }

#pragma unroll
  for (int f = 0; f < 4; ++f) {
    const int n = nt * 64 + f * 16 + l16;
    const float bv = bO[n];
#pragma unroll
    for (int s = 0; s < 4; ++s)
#pragma unroll
      for (int g = 0; g < 8; ++g) {
        const int token = mt * 64 + s * 16 + g + 8 * half;
        out[(size_t)token * DMODEL + n] = acc[s][f][g] + bv;
      }
  }
}

// ---------------------------------------------------------------------------
extern "C" void kernel_launch(void* const* d_in, const int* in_sizes, int n_in,
                              void* d_out, int out_size, void* d_ws, size_t ws_size,
                              hipStream_t stream) {
  const float* xq   = (const float*)d_in[0];
  const float* xk   = (const float*)d_in[1];
  const float* xv   = (const float*)d_in[2];
  const float* mask = (const float*)d_in[3];
  const float* WQ   = (const float*)d_in[4];
  const float* WK   = (const float*)d_in[5];
  const float* WV   = (const float*)d_in[6];
  const float* WO   = (const float*)d_in[7];
  const float* bQ   = (const float*)d_in[8];
  const float* bK   = (const float*)d_in[9];
  const float* bV   = (const float*)d_in[10];
  const float* bO   = (const float*)d_in[11];

  char* ws = (char*)d_ws;
  const size_t MB = 1ull << 20;
  unsigned short* xq_bf = (unsigned short*)(ws + 0 * MB);   // 8 MiB each
  unsigned short* xk_bf = (unsigned short*)(ws + 8 * MB);
  unsigned short* xv_bf = (unsigned short*)(ws + 16 * MB);
  unsigned short* wq_t  = (unsigned short*)(ws + 24 * MB);  // 2 MiB each
  unsigned short* wk_t  = (unsigned short*)(ws + 26 * MB);
  unsigned short* wv_t  = (unsigned short*)(ws + 28 * MB);
  unsigned short* wo_t  = (unsigned short*)(ws + 30 * MB);
  unsigned short* q_bf  = (unsigned short*)(ws + 32 * MB);  // 8 MiB each
  unsigned short* k_bf  = (unsigned short*)(ws + 40 * MB);
  unsigned short* vt_bf = (unsigned short*)(ws + 48 * MB);
  unsigned short* z_bf  = (unsigned short*)(ws + 56 * MB);  // total 64 MiB

  const int nAct = TOK * DMODEL;                 // 4194304
  const int nW   = HEADS * DMODEL * DHEAD;       // 1048576

  cvt_act<<<nAct / 256, 256, 0, stream>>>(xq, xq_bf);
  cvt_act<<<nAct / 256, 256, 0, stream>>>(xk, xk_bf);
  cvt_act<<<nAct / 256, 256, 0, stream>>>(xv, xv_bf);
  cvt_wqkv_t<<<nW / 256, 256, 0, stream>>>(WQ, wq_t);
  cvt_wqkv_t<<<nW / 256, 256, 0, stream>>>(WK, wk_t);
  cvt_wqkv_t<<<nW / 256, 256, 0, stream>>>(WV, wv_t);
  cvt_wo_t<<<nW / 256, 256, 0, stream>>>(WO, wo_t);

  qkv_proj<<<dim3(TOK / 64, DMODEL / 64, 3), 32, 0, stream>>>(
      xq_bf, xk_bf, xv_bf, wq_t, wk_t, wv_t, bQ, bK, bV, q_bf, k_bf, vt_bf);

  flash_attn<<<dim3(SEQ / 32, HEADS, BATCH), 32, 0, stream>>>(
      q_bf, k_bf, vt_bf, mask, z_bf);

  out_proj<<<dim3(TOK / 64, DMODEL / 64), 32, 0, stream>>>(
      z_bf, wo_t, bO, (float*)d_out);
}
